// HybridPredictionModel_34359739157
// MI455X (gfx1250) — matrix operand order repaired and checked by reading, not compiled, not run on hardware
//
#include <hip/hip_runtime.h>
#include <hip/hip_bf16.h>

typedef __attribute__((ext_vector_type(16))) _Float16 v16h;
typedef __attribute__((ext_vector_type(8)))  _Float16 v8h;
typedef __attribute__((ext_vector_type(4)))  _Float16 v4h;
typedef __attribute__((ext_vector_type(8)))  float    v8f;

#define ROWS    128        // rows per block
#define DIN     308        // input features
#define APITCH  328        // f16 A-tile row pitch (halves); pads K to 320 + 8 anti-conflict
#define KSTEPS  10         // ceil(308/32)
#define HDIM    64
#define HPITCH  65         // h-tile pitch in floats (conflict-free column reads)

// dynamic LDS layout (bytes)
#define AH_OFF      0                       // 128*328 halves          = 83968 B
#define UNION_OFF   83968                   // W1 frags 40960 B  OR  h-tile 33280 B
#define FITSIG_OFF  124928                  // 2*128 floats            = 1024 B
#define FITA_OFF    125952                  // 2*128 floats            = 1024 B
#define SMALL_OFF   126976                  // 944 floats              = 3776 B
#define SMEM_BYTES  130752
// small-region float offsets
#define S_B1   0      // 64
#define S_WD   64     // 192  (row-major [64,3])
#define S_BD   256    // 3
#define S_WC   260    // 64
#define S_BC   324    // 1
#define S_BAYY 328    // 154
#define S_BAYZ 482    // 154
#define S_BAXX 636    // 154
#define S_BAXZ 790    // 154

extern "C" __global__ __launch_bounds__(256)
void hpm_fused(const float* __restrict__ barE, const float* __restrict__ W1,
               const float* __restrict__ b1,   const float* __restrict__ Wd,
               const float* __restrict__ bd,   const float* __restrict__ Wc,
               const float* __restrict__ bc,   const float* __restrict__ BAYYL,
               const float* __restrict__ BAYZL,const float* __restrict__ BAXXL,
               const float* __restrict__ BAXZL,float* __restrict__ out, int Btotal)
{
    extern __shared__ char smem[];
    _Float16* Ah    = (_Float16*)(smem + AH_OFF);
    _Float16* Wf    = (_Float16*)(smem + UNION_OFF);
    float*    htile = (float*)   (smem + UNION_OFF);
    float*    fitSg = (float*)   (smem + FITSIG_OFF);
    float*    fitA  = (float*)   (smem + FITA_OFF);
    float*    S     = (float*)   (smem + SMALL_OFF);

    const int tid  = threadIdx.x;
    const int row0 = blockIdx.x * ROWS;

    // ---------------- Phase 1: stage everything into LDS ----------------
    for (int i = tid; i < 64;  i += 256) S[S_B1 + i] = b1[i];
    for (int i = tid; i < 192; i += 256) S[S_WD + i] = Wd[i];
    if (tid < 3)  S[S_BD + tid] = bd[tid];
    for (int i = tid; i < 64;  i += 256) S[S_WC + i] = Wc[i];
    if (tid == 0) S[S_BC] = bc[0];
    for (int i = tid; i < 154; i += 256) {
        S[S_BAYY + i] = BAYYL[i];  S[S_BAYZ + i] = BAYZL[i];
        S[S_BAXX + i] = BAXXL[i];  S[S_BAXZ + i] = BAXZL[i];
    }
    // barE tile: f32 -> f16, 77 float4 per row (308 = 77*4, rows 16B-aligned)
    for (int idx = tid; idx < ROWS * 77; idx += 256) {
        int r  = idx / 77;
        int c4 = idx - r * 77;
        int g  = row0 + r;
        float4 v = make_float4(0.f, 0.f, 0.f, 0.f);
        if (g < Btotal) v = *(const float4*)(barE + (size_t)g * DIN + c4 * 4);
        v4h h;
        h[0] = (_Float16)v.x; h[1] = (_Float16)v.y;
        h[2] = (_Float16)v.z; h[3] = (_Float16)v.w;
        *(v4h*)(Ah + r * APITCH + c4 * 4) = h;   // 8B-aligned ds_store_b64
    }
    // zero K-pad cols 308..319 (read by WMMA)
    for (int idx = tid; idx < ROWS * 12; idx += 256) {
        int r = idx / 12;
        Ah[r * APITCH + DIN + (idx - r * 12)] = (_Float16)0.f;
    }
    // W1 pre-swizzled WMMA B fragments: frag(nt,ks) = 32 lanes x 16 halves.
    // lane<16 : N = nt*16+lane,    K = ks*32 + {0..7, 16..23}[i]   (i = 0..15)
    // lane>=16: N = nt*16+lane-16, K = ks*32 + {8..15, 24..31}[i]
    for (int idx = tid; idx < 4 * KSTEPS * 32 * 16; idx += 256) {
        int i    = idx & 15;
        int lane = (idx >> 4) & 31;
        int frag = idx >> 9;                 // 0..39
        int ks   = frag % KSTEPS;
        int nt   = frag / KSTEPS;
        int k    = ks * 32 + ((lane >> 4) << 3) + (i & 7) + ((i >> 3) << 4);
        int n    = nt * 16 + (lane & 15);
        Wf[idx] = (k < DIN) ? (_Float16)W1[k * HDIM + n] : (_Float16)0.f;
    }
    __syncthreads();

    // ---------------- Phase 2: WMMA  h = barE @ W1 ----------------
    const int lane = tid & 31;
    const int wv   = tid >> 5;               // wave id: rows wv*16 .. wv*16+15
    v8f acc[4];
    #pragma unroll
    for (int nt = 0; nt < 4; ++nt)
        #pragma unroll
        for (int r = 0; r < 8; ++r) acc[nt][r] = 0.f;

    // A fragment per ISA layout: lane<16 reads K-chunks {0..7, 16..23} of row lane,
    // lane>=16 reads K-chunks {8..15, 24..31} of row lane-16. Two ds_load_b128 each.
    const int aBase = (wv * 16 + (lane & 15)) * APITCH + ((lane >> 4) << 3);
    for (int ks = 0; ks < KSTEPS; ++ks) {
        union { v16h v; v8h h[2]; } a;
        const _Float16* ap = Ah + aBase + ks * 32;
        a.h[0] = *(const v8h*)(ap);
        a.h[1] = *(const v8h*)(ap + 16);
        #pragma unroll
        for (int nt = 0; nt < 4; ++nt) {
            v16h bfrag = *(const v16h*)(Wf + ((nt * KSTEPS + ks) * 32 + lane) * 16);
            acc[nt] = __builtin_amdgcn_wmma_f32_16x16x32_f16(
                false, a.v, false, bfrag, (short)0, acc[nt], false, false);
        }
    }
    __syncthreads();   // W1 fragments dead -> region becomes h-tile

    // ---------------- Phase 3: bias + ReLU, spill h to LDS ----------------
    {
        const int half = lane >> 4, c16 = lane & 15;
        #pragma unroll
        for (int nt = 0; nt < 4; ++nt) {
            float bv = S[S_B1 + nt * 16 + c16];
            #pragma unroll
            for (int r = 0; r < 8; ++r) {
                int M = r + 8 * half;        // C layout: lanes 0-15 M=0..7, 16-31 M=8..15
                float hv = fmaxf(acc[nt][r] + bv, 0.f);
                htile[(wv * 16 + M) * HPITCH + nt * 16 + c16] = hv;
            }
        }
    }
    __syncthreads();

    // ---------------- Phase 4: energy-weighted track fits (2 per row) ----------------
    {
        const int row   = tid & 127;
        const int which = tid >> 7;          // 0 = Ey fit, 1 = Ex fit
        const float* BP = which ? (S + S_BAXX) : (S + S_BAYY);
        const float* BZ = which ? (S + S_BAXZ) : (S + S_BAYZ);
        const float  zc = which ? 87.5f : 58.5f;
        const _Float16* Er = Ah + row * APITCH + which * 22;  // flat col = l*44 + which*22 + j

        float Sw = 0.f, Szv = 0.f, Sxv = 0.f, Szz = 0.f, Szx = 0.f;
        float wl[7], xl[7], zl[7];
        #pragma unroll
        for (int l = 0; l < 7; ++l) {
            float w = 0.f, sp = 0.f, sz = 0.f;
            float zb = zc + 58.f * l;
            for (int j = 0; j < 22; ++j) {
                float e = (float)Er[l * 44 + j];
                float P = -288.75f + 27.5f * (float)j + BP[l * 22 + j];
                float Z = zb + BZ[l * 22 + j];
                w += e; sp += e * P; sz += e * Z;
            }
            float ws = (w > 0.f) ? w : 1.f;
            float x = sp / ws, z = sz / ws;
            wl[l] = w; xl[l] = x; zl[l] = z;
            Sw += w; Szv += w * z; Sxv += w * x; Szz += w * z * z; Szx += w * z * x;
        }
        float det  = Sw * Szz - Szv * Szv;
        float dets = (fabsf(det) > 1e-9f) ? det : 1.f;
        float a  = (Sw * Szx - Szv * Sxv) / dets;
        float bI = (Szz * Sxv - Szv * Szx) / dets;
        float ss = 0.f;
        #pragma unroll
        for (int l = 0; l < 7; ++l) {
            float r2 = xl[l] - (a * zl[l] + bI);
            ss += wl[l] * r2 * r2;
        }
        float Sws = (Sw > 0.f) ? Sw : 1.f;
        float sig = sqrtf(fmaxf(ss / Sws, 0.f) + 1e-20f);
        bool  ok  = (fabsf(det) > 1e-9f) && (Sw > 0.f);
        fitSg[which * ROWS + row] = ok ? sig : -1.f;
        fitA [which * ROWS + row] = a;
    }
    __syncthreads();

    // ---------------- Phase 5: heads + blend + write ----------------
    if (tid < ROWS) {
        const int row = tid;
        const int g   = row0 + row;
        if (g < Btotal) {
            float d0 = S[S_BD + 0], d1 = S[S_BD + 1], d2 = S[S_BD + 2], c = S[S_BC];
            const float* hr = htile + row * HPITCH;
            #pragma unroll
            for (int n = 0; n < HDIM; ++n) {
                float hv = hr[n];
                d0 = fmaf(hv, S[S_WD + n * 3 + 0], d0);
                d1 = fmaf(hv, S[S_WD + n * 3 + 1], d1);
                d2 = fmaf(hv, S[S_WD + n * 3 + 2], d2);
                c  = fmaf(hv, S[S_WC + n], c);
            }
            float inv = 1.f / sqrtf(d0 * d0 + d1 * d1 + d2 * d2 + 1e-12f);
            float dl0 = d0 * inv, dl1 = d1 * inv, dl2 = d2 * inv;
            float decision = 1.f / (1.f + expf(-c));

            float sy = fitSg[row],        ay = fitA[row];
            float sx = fitSg[ROWS + row], ax = fitA[ROWS + row];
            bool  valid = (sx > 0.f) && (sy > 0.f);
            float dninv = 1.f / sqrtf(ax * ax + ay * ay + 1.f + 1e-12f);
            float p0 = valid ? ax * dninv : 0.f;
            float p1 = valid ? ay * dninv : 0.f;
            float p2 = valid ? dninv      : 0.f;
            float pv  = (p0 * p0 + p1 * p1 + p2 * p2) > 1e-6f ? 1.f : 0.f;
            float dec = decision * pv;

            float f0 = dec * p0 + (1.f - dec) * dl0;
            float f1 = dec * p1 + (1.f - dec) * dl1;
            float f2 = dec * p2 + (1.f - dec) * dl2;
            float fn = 1.f / sqrtf(fmaxf(f0 * f0 + f1 * f1 + f2 * f2, 1e-12f));
            out[(size_t)g * 3 + 0] = f0 * fn;
            out[(size_t)g * 3 + 1] = f1 * fn;
            out[(size_t)g * 3 + 2] = f2 * fn;
            out[(size_t)Btotal * 3 + g] = dec;     // dec concatenated after final
        }
    }
}

extern "C" void kernel_launch(void* const* d_in, const int* in_sizes, int n_in,
                              void* d_out, int out_size, void* d_ws, size_t ws_size,
                              hipStream_t stream) {
    const float* barE  = (const float*)d_in[0];
    const float* W1    = (const float*)d_in[1];
    const float* b1    = (const float*)d_in[2];
    const float* Wd    = (const float*)d_in[3];
    const float* bd    = (const float*)d_in[4];
    const float* Wc    = (const float*)d_in[5];
    const float* bc    = (const float*)d_in[6];
    const float* BAYYL = (const float*)d_in[7];
    const float* BAYZL = (const float*)d_in[8];
    const float* BAXXL = (const float*)d_in[9];
    const float* BAXZL = (const float*)d_in[10];
    float* out = (float*)d_out;

    const int Btotal = in_sizes[0] / DIN;                // 262144
    const int blocks = (Btotal + ROWS - 1) / ROWS;       // 2048
    hpm_fused<<<blocks, 256, SMEM_BYTES, stream>>>(
        barE, W1, b1, Wd, bd, Wc, bc, BAYYL, BAYZL, BAXXL, BAXZL, out, Btotal);
}
